// GNN_node_6640019440405
// MI455X (gfx1250) — matrix operand order repaired and checked by reading, not compiled, not run on hardware
//
#include <hip/hip_runtime.h>
#include <hip/hip_bf16.h>
#include <math.h>

// ---------------------------------------------------------------------------
// CDNA5 (gfx1250) GNN pipeline, round 3 (polish).
// MLPs computed transposed (Y^T = W2^T * leaky(W1^T * X^T)) so that:
//   A fragments = weight tiles from LDS, contiguous 16B chunks (ds_load_b128)
//   B fragments = activation rows, contiguous global float4 loads + cvt
//   C/D tiles   = [out_feat][row] -> 8 contiguous floats per lane on store
// LeakyReLU as max(x, slope*x) -> single v_max_num_f32 (slope < 1).
// psi-MLP over 1M edges fused with gather (hn[snk_net]) and atomic
// scatter-add (node_agg[snk_n]).  Scatter segsum/segmax via float atomics.
// ---------------------------------------------------------------------------

#define NEG_SLOPE 0.01f

typedef __attribute__((ext_vector_type(16))) _Float16 v16h;
typedef __attribute__((ext_vector_type(8)))  _Float16 v8h;
typedef __attribute__((ext_vector_type(8)))  float    v8f;
typedef __attribute__((ext_vector_type(4)))  float    v4f;

__device__ __forceinline__ float leaky(float x) {
  return fmaxf(x, NEG_SLOPE * x);  // exact for 0 < slope < 1
}

__device__ __forceinline__ v8f wmma16x16x32(v16h a, v16h b, v8f c) {
  // D = A(16x32,f16) * B(32x16,f16) + C(16x16,f32)
  return __builtin_amdgcn_wmma_f32_16x16x32_f16(
      /*neg_a=*/false, a, /*neg_b=*/false, b,
      /*c_mod=*/(short)0, c, /*reuse_a=*/false, /*reuse_b=*/false);
}

__device__ __forceinline__ void atomicMaxFloat(float* addr, float val) {
  int* ia = (int*)addr;
  int old = __float_as_int(*addr);
  while (__int_as_float(old) < val) {
    int assumed = old;
    old = atomicCAS(ia, assumed, __float_as_int(val));
    if (old == assumed) break;
  }
}

__device__ __forceinline__ void atomicMinFloat(float* addr, float val) {
  int* ia = (int*)addr;
  int old = __float_as_int(*addr);
  while (__int_as_float(old) > val) {
    int assumed = old;
    old = atomicCAS(ia, assumed, __float_as_int(val));
    if (old == assumed) break;
  }
}

// Load one 16-wide K-chunk of the (concat(X0[g0],X1[g1]) zero-padded) row as
// f16.  Chunk base k0 is always a multiple of 16.
template<int D0, int D1>
__device__ __forceinline__ v16h load_x_chunk(const float* __restrict__ X0, long r0,
                                             const float* __restrict__ X1, long r1,
                                             int k0, bool valid) {
  constexpr int DIN = D0 + D1;
  v16h out;
  if (!valid) {
    #pragma unroll
    for (int e = 0; e < 16; ++e) out[e] = (_Float16)0.0f;
  } else if (k0 + 16 <= D0) {
    const v4f* p = (const v4f*)(X0 + r0 * D0 + k0);
    #pragma unroll
    for (int j = 0; j < 4; ++j) {
      v4f f = p[j];
      #pragma unroll
      for (int t = 0; t < 4; ++t) out[4 * j + t] = (_Float16)f[t];
    }
  } else if (D1 > 0 && k0 >= D0 && k0 + 16 <= DIN) {
    const v4f* p = (const v4f*)(X1 + r1 * D1 + (k0 - D0));
    #pragma unroll
    for (int j = 0; j < 4; ++j) {
      v4f f = p[j];
      #pragma unroll
      for (int t = 0; t < 4; ++t) out[4 * j + t] = (_Float16)f[t];
    }
  } else {  // chunk straddles a boundary (psi: k=64 -> eattr, 65.. -> pad)
    #pragma unroll
    for (int e = 0; e < 16; ++e) {
      int k = k0 + e;
      float v = 0.0f;
      if (k < D0)                 v = X0[r0 * D0 + k];
      else if (D1 > 0 && k < DIN) v = X1[r1 * D1 + (k - D0)];
      out[e] = (_Float16)v;
    }
  }
  return out;
}

// Weight A-fragment from LDS laid out [out_feat][Kstride] (contiguous K).
__device__ __forceinline__ v16h load_w_frag(const _Float16* __restrict__ w,
                                            int row, int kstride, int kt, int hi) {
  const _Float16* p = w + (long)row * kstride + kt * 32 + hi * 16;
  v8h lo = *(const v8h*)p;          // e=0..7  : k = kt*32 + 16*hi + e
  v8h hh = *(const v8h*)(p + 8);    // e=8..15 : k = kt*32 + 16*hi + e
  return __builtin_shufflevector(lo, hh, 0, 1, 2, 3, 4, 5, 6, 7,
                                 8, 9, 10, 11, 12, 13, 14, 15);
}

// ---------------------------------------------------------------------------
// Fused 2-layer MLP:  Y = leaky(concat(X0[g0],X1[g1]) @ W1 + b1) @ W2 + b2
// Optional residual add, optional atomic scatter-add on output rows.
// Block = 128 threads = 4 waves; each wave owns 16 rows (64 rows/block).
// ---------------------------------------------------------------------------
template<int D0, int D1, int DH, int DOUT>
__launch_bounds__(128)
__global__ void mlp2_wmma_kernel(const float* __restrict__ X0, const int* __restrict__ g0,
                                 const float* __restrict__ X1, const int* __restrict__ g1,
                                 const float* __restrict__ W1, const float* __restrict__ B1,
                                 const float* __restrict__ W2, const float* __restrict__ B2,
                                 float* __restrict__ Y, const int* __restrict__ scat,
                                 const float* __restrict__ resid, int nrows)
{
  constexpr int WAVES = 4;
  constexpr int DIN   = D0 + D1;
  constexpr int KPAD  = (DIN + 31) & ~31;
  constexpr int KT1   = KPAD / 32;
  constexpr int NT1   = DH / 16;
  constexpr int KT2   = DH / 32;
  constexpr int DOUTP = (DOUT + 15) & ~15;
  constexpr int NT2   = DOUTP / 16;
  constexpr int WSZ   = (DH * KPAD > DOUTP * DH) ? DH * KPAD : DOUTP * DH;

  __shared__ _Float16 wsh[WSZ];              // W1^T then (after barrier) W2^T
  __shared__ _Float16 hbuf[WAVES * 16 * DH]; // hidden acts, [row][feat], f16
  __shared__ float    b1s[DH];
  __shared__ float    b2s[DOUTP];

  const int tid  = threadIdx.x;
  const int lane = tid & 31;
  const int wave = tid >> 5;
  const int hi   = (lane >> 4) & 1;
  const int col  = lane & 15;

  // ---- stage W1^T into LDS as [hidden][KPAD], f16, K zero-padded ----
  for (int i = tid; i < DH * KPAD; i += 128) {
    int n = i / KPAD, k = i % KPAD;
    float v = (k < DIN) ? W1[k * DH + n] : 0.0f;
    wsh[i] = (_Float16)v;
  }
  for (int i = tid; i < DH; i += 128) b1s[i] = B1[i];
  __syncthreads();

  const int  rowBase = (blockIdx.x * WAVES + wave) * 16;
  const int  myRow   = rowBase + col;
  const bool valid   = (myRow < nrows);
  long r0 = 0, r1 = 0;
  if (valid) {
    r0 = g0 ? (long)g0[myRow] : (long)myRow;
    if (D1 > 0) r1 = g1 ? (long)g1[myRow] : (long)myRow;
  }
  if (valid && resid) __builtin_prefetch(&resid[(long)myRow * DOUT], 0, 1);

  // ---- B fragments: this lane's input row, 16 contiguous K per fragment ----
  v16h xb[KT1];
  #pragma unroll
  for (int kt = 0; kt < KT1; ++kt)
    xb[kt] = load_x_chunk<D0, D1>(X0, r0, X1, r1, kt * 32 + hi * 16, valid);

  // ---- stage 1: H^T tiles = W1^T x X^T; pack row slice to hbuf ----
  _Float16* hrow = &hbuf[(wave * 16 + col) * DH];  // this lane's row slice
  #pragma unroll
  for (int nt = 0; nt < NT1; ++nt) {
    v8f c = {};
    #pragma unroll
    for (int kt = 0; kt < KT1; ++kt)
      c = wmma16x16x32(load_w_frag(wsh, nt * 16 + col, KPAD, kt, hi), xb[kt], c);
    // features h = nt*16 + hi*8 + r for this lane's row (= col)
    v4f bA = *(const v4f*)&b1s[nt * 16 + hi * 8];
    v4f bB = *(const v4f*)&b1s[nt * 16 + hi * 8 + 4];
    v8h packed;
    #pragma unroll
    for (int r = 0; r < 8; ++r) {
      float v = c[r] + ((r < 4) ? bA[r] : bB[r - 4]);
      packed[r] = (_Float16)leaky(v);
    }
    *(v8h*)&hrow[nt * 16 + hi * 8] = packed;  // one 16B ds_store
  }

  __syncthreads();  // all waves done with W1 region
  // ---- stage W2^T into LDS as [out][DH], out zero-padded ----
  for (int i = tid; i < DOUTP * DH; i += 128) {
    int o = i / DH, k = i % DH;
    float v = (o < DOUT) ? W2[k * DOUT + o] : 0.0f;
    wsh[i] = (_Float16)v;
  }
  for (int i = tid; i < DOUTP; i += 128) b2s[i] = (i < DOUT) ? B2[i] : 0.0f;
  __syncthreads();

  // ---- B fragments from hbuf: contiguous 32B per fragment ----
  v16h hb[KT2];
  #pragma unroll
  for (int kt = 0; kt < KT2; ++kt)
    hb[kt] = *(const v16h*)&hrow[kt * 32 + hi * 16];

  // ---- stage 2 + epilogue ----
  #pragma unroll
  for (int nt = 0; nt < NT2; ++nt) {
    v8f c = {};
    #pragma unroll
    for (int kt = 0; kt < KT2; ++kt)
      c = wmma16x16x32(load_w_frag(wsh, nt * 16 + col, DH, kt, hi), hb[kt], c);
    const int o0   = nt * 16 + hi * 8;      // 8 contiguous out features
    const int grow = rowBase + col;
    if (grow < nrows && o0 < DOUT) {
      v4f bA = *(const v4f*)&b2s[o0];
      v4f bB = *(const v4f*)&b2s[o0 + 4];
      float vals[8];
      #pragma unroll
      for (int r = 0; r < 8; ++r) vals[r] = c[r] + ((r < 4) ? bA[r] : bB[r - 4]);
      long base = (long)grow * DOUT + o0;
      if (resid) {
        v4f rA = *(const v4f*)&resid[base];
        v4f rB = *(const v4f*)&resid[base + 4];
        #pragma unroll
        for (int r = 0; r < 8; ++r) vals[r] += (r < 4) ? rA[r] : rB[r - 4];
      }
      if (scat) {
        long sb = (long)scat[grow] * DOUT + o0;
        #pragma unroll
        for (int r = 0; r < 8; ++r) atomicAdd(&Y[sb + r], vals[r]);
      } else {
        v4f sA, sB;
        #pragma unroll
        for (int r = 0; r < 4; ++r) { sA[r] = vals[r]; sB[r] = vals[r + 4]; }
        *(v4f*)&Y[base]     = sA;   // global_store_b128
        *(v4f*)&Y[base + 4] = sB;
      }
    }
  }
}

// ---------------------------------------------------------------------------
// Support kernels
// ---------------------------------------------------------------------------
__global__ void compute_idx_kernel(const int* __restrict__ pos, int* __restrict__ idx, int n) {
  int i = blockIdx.x * blockDim.x + threadIdx.x;
  if (i < n) idx[i] = pos[2 * i] * 512 + pos[2 * i + 1];
}

__global__ void fill_f32_kernel(float* __restrict__ p, float v, long n) {
  long i = (long)blockIdx.x * blockDim.x + threadIdx.x;
  long s = (long)gridDim.x * blockDim.x;
  for (; i < n; i += s) p[i] = v;
}

__global__ void fill_from_scalar_kernel(float* __restrict__ p, const float* __restrict__ sc, long n) {
  float v = *sc;
  long i = (long)blockIdx.x * blockDim.x + threadIdx.x;
  long s = (long)gridDim.x * blockDim.x;
  for (; i < n; i += s) p[i] = v;
}

__global__ void min_reduce_kernel(const float* __restrict__ x, long n, float* __restrict__ out) {
  long i = (long)blockIdx.x * blockDim.x + threadIdx.x;
  long s = (long)gridDim.x * blockDim.x;
  float m = 3.402823466e38f;
  for (; i < n; i += s) m = fminf(m, x[i]);
  #pragma unroll
  for (int d = 16; d; d >>= 1) m = fminf(m, __shfl_xor(m, d, 32));
  if ((threadIdx.x & 31) == 0) atomicMinFloat(out, m);
}

__global__ void scatter_add_kernel(const float* __restrict__ x, const int* __restrict__ idx,
                                   float* __restrict__ out, long n, int C) {
  long t = (long)blockIdx.x * blockDim.x + threadIdx.x;
  if (t >= n * C) return;
  long i = t / C; int c = (int)(t - i * C);
  atomicAdd(&out[(long)idx[i] * C + c], x[t]);
}

__global__ void scatter_max_kernel(const float* __restrict__ x, const int* __restrict__ idx,
                                   float* __restrict__ out, long n, int C) {
  long t = (long)blockIdx.x * blockDim.x + threadIdx.x;
  if (t >= n * C) return;
  long i = t / C; int c = (int)(t - i * C);
  atomicMaxFloat(&out[(long)idx[i] * C + c], x[t]);
}

__global__ void edge_gather_add_kernel(const float* __restrict__ src, const int* __restrict__ sidx,
                                       const int* __restrict__ didx, float* __restrict__ out,
                                       long nE, int C) {
  long t = (long)blockIdx.x * blockDim.x + threadIdx.x;
  if (t >= nE * C) return;
  long e = t / C; int c = (int)(t - e * C);
  atomicAdd(&out[(long)didx[e] * C + c], src[(long)sidx[e] * C + c]);
}

// LayerNorm(64) + leaky relu, one wave32 per row (2 cols/lane).
__global__ void ln_leaky_kernel(float* __restrict__ X, const float* __restrict__ g,
                                const float* __restrict__ b, int rows) {
  int  lane = threadIdx.x & 31;
  long row  = ((long)blockIdx.x * blockDim.x + threadIdx.x) >> 5;
  if (row >= rows) return;
  float x0 = X[row * 64 + lane];
  float x1 = X[row * 64 + 32 + lane];
  float s = x0 + x1;
  #pragma unroll
  for (int d = 16; d; d >>= 1) s += __shfl_xor(s, d, 32);
  float mu = s * (1.0f / 64.0f);
  float d0 = x0 - mu, d1 = x1 - mu;
  float v = d0 * d0 + d1 * d1;
  #pragma unroll
  for (int d = 16; d; d >>= 1) v += __shfl_xor(v, d, 32);
  float rs = rsqrtf(v * (1.0f / 64.0f) + 1e-5f);
  X[row * 64 + lane]      = leaky(d0 * rs * g[lane]      + b[lane]);
  X[row * 64 + 32 + lane] = leaky(d1 * rs * g[32 + lane] + b[32 + lane]);
}

// ---------------------------------------------------------------------------
// Host orchestration
// ---------------------------------------------------------------------------
extern "C" void kernel_launch(void* const* d_in, const int* in_sizes, int n_in,
                              void* d_out, int out_size, void* d_ws, size_t ws_size,
                              hipStream_t stream) {
  const int  N = 300000, M = 250000, E = 1000000, NS = 512 * 512;
  const float FMAX = 3.402823466e38f;

  auto F = [&](int i) { return (const float*)d_in[i]; };
  const float* h_inst = F(0);
  const float* h_net  = F(1);
  const int*   ei_s2n = (const int*)d_in[2];  // [2,M]: row0=src node, row1=net
  const int*   ei_k2n = (const int*)d_in[3];  // [2,E]: row0=sink node, row1=net
  const float* eattr  = F(4);                 // [E,1]
  const int*   pos    = (const int*)d_in[5];  // [N,2]
  float*       out    = (float*)d_out;

  // Workspace carve (256B aligned slabs)
  char* wsp = (char*)d_ws;
  auto alloc = [&](size_t bytes) {
    void* p = (void*)wsp;
    wsp += (bytes + 255) & ~(size_t)255;
    return p;
  };
  int*   idx  = (int*)  alloc((size_t)N * 4);
  float* minv = (float*)alloc(256);
  float* h    = (float*)alloc((size_t)N  * 64 * 4);
  float* h2   = (float*)alloc((size_t)N  * 64 * 4);
  float* nb   = (float*)alloc((size_t)N  * 64 * 4);   // node_agg
  float* hnA  = (float*)alloc((size_t)M  * 64 * 4);
  float* hnB  = (float*)alloc((size_t)M  * 64 * 4);
  float* na   = (float*)alloc((size_t)M  * 64 * 4);   // net_agg
  float* vn   = (float*)alloc((size_t)NS * 64 * 4);
  float* us   = (float*)alloc((size_t)NS * 64 * 4);   // util sum
  float* um   = (float*)alloc((size_t)NS * 64 * 4);   // util max
  float* hn_cur = hnA, *hn_nxt = hnB;

  // instance -> site index
  compute_idx_kernel<<<(N + 255) / 256, 256, 0, stream>>>(pos, idx, N);

  // raw util maps of h_inst [N,16] -> [NS,16] each
  fill_f32_kernel<<<2048, 256, 0, stream>>>(us, 0.0f, (long)NS * 16);
  fill_f32_kernel<<<1, 32, 0, stream>>>(minv, FMAX, 1);
  min_reduce_kernel<<<1024, 256, 0, stream>>>(h_inst, (long)N * 16, minv);
  fill_from_scalar_kernel<<<2048, 256, 0, stream>>>(um, minv, (long)NS * 16);
  scatter_add_kernel<<<(int)(((long)N * 16 + 255) / 256), 256, 0, stream>>>(h_inst, idx, us, N, 16);
  scatter_max_kernel<<<(int)(((long)N * 16 + 255) / 256), 256, 0, stream>>>(h_inst, idx, um, N, 16);

  // encoders
  mlp2_wmma_kernel<16, 16, 128, 64><<<(NS + 63) / 64, 128, 0, stream>>>(
      us, nullptr, um, nullptr, F(14), F(15), F(16), F(17), vn, nullptr, nullptr, NS);
  mlp2_wmma_kernel<16, 0, 64, 64><<<(N + 63) / 64, 128, 0, stream>>>(
      h_inst, nullptr, nullptr, nullptr, F(6), F(7), F(8), F(9), h, nullptr, nullptr, N);
  mlp2_wmma_kernel<16, 0, 64, 64><<<(M + 63) / 64, 128, 0, stream>>>(
      h_net, nullptr, nullptr, nullptr, F(10), F(11), F(12), F(13), hn_cur, nullptr, nullptr, M);

  for (int l = 0; l < 3; ++l) {
    const int LB = 18 + 22 * l;  // back(4) phi(4) psi(4) mlp(4) ln(2) mlp_vn(4)

    // h = back(concat(h, vn[idx]))
    mlp2_wmma_kernel<64, 64, 128, 64><<<(N + 63) / 64, 128, 0, stream>>>(
        h, nullptr, vn, idx, F(LB + 0), F(LB + 1), F(LB + 2), F(LB + 3), h2, nullptr, nullptr, N);

    // net_agg[net] += h[src]
    fill_f32_kernel<<<4096, 256, 0, stream>>>(na, 0.0f, (long)M * 64);
    edge_gather_add_kernel<<<(int)(((long)M * 64 + 255) / 256), 256, 0, stream>>>(
        h2, ei_s2n, ei_s2n + M, na, M, 64);

    // hn = phi(concat(hn, net_agg))
    mlp2_wmma_kernel<64, 64, 64, 64><<<(M + 63) / 64, 128, 0, stream>>>(
        hn_cur, nullptr, na, nullptr, F(LB + 4), F(LB + 5), F(LB + 6), F(LB + 7),
        hn_nxt, nullptr, nullptr, M);

    // node_agg[sink] += psi(concat(hn[net], eattr))  (gather + scatter fused)
    fill_f32_kernel<<<4096, 256, 0, stream>>>(nb, 0.0f, (long)N * 64);
    mlp2_wmma_kernel<64, 1, 64, 64><<<(E + 63) / 64, 128, 0, stream>>>(
        hn_nxt, ei_k2n + E, eattr, nullptr, F(LB + 8), F(LB + 9), F(LB + 10), F(LB + 11),
        nb, ei_k2n, nullptr, E);

    // h = mlp(concat(h, node_agg)) + h
    mlp2_wmma_kernel<64, 64, 64, 64><<<(N + 63) / 64, 128, 0, stream>>>(
        h2, nullptr, nb, nullptr, F(LB + 12), F(LB + 13), F(LB + 14), F(LB + 15),
        h, nullptr, h2, N);

    // shared LN + leaky
    ln_leaky_kernel<<<(int)(((long)N * 32 + 255) / 256), 256, 0, stream>>>(h, F(LB + 16), F(LB + 17), N);
    ln_leaky_kernel<<<(int)(((long)M * 32 + 255) / 256), 256, 0, stream>>>(hn_nxt, F(LB + 16), F(LB + 17), M);
    { float* t = hn_cur; hn_cur = hn_nxt; hn_nxt = t; }

    // util maps of new h [N,64]
    fill_f32_kernel<<<4096, 256, 0, stream>>>(us, 0.0f, (long)NS * 64);
    fill_f32_kernel<<<1, 32, 0, stream>>>(minv, FMAX, 1);
    min_reduce_kernel<<<2048, 256, 0, stream>>>(h, (long)N * 64, minv);
    fill_from_scalar_kernel<<<4096, 256, 0, stream>>>(um, minv, (long)NS * 64);
    scatter_add_kernel<<<(int)(((long)N * 64 + 255) / 256), 256, 0, stream>>>(h, idx, us, N, 64);
    scatter_max_kernel<<<(int)(((long)N * 64 + 255) / 256), 256, 0, stream>>>(h, idx, um, N, 64);

    // vn = mlp_vn(concat(us, um)) + vn
    mlp2_wmma_kernel<64, 64, 128, 64><<<(NS + 63) / 64, 128, 0, stream>>>(
        us, nullptr, um, nullptr, F(LB + 18), F(LB + 19), F(LB + 20), F(LB + 21),
        vn, nullptr, vn, NS);
  }

  // output heads (DOUT=8, padded to 16 inside the kernel)
  mlp2_wmma_kernel<64, 0, 128, 8><<<(N + 63) / 64, 128, 0, stream>>>(
      h, nullptr, nullptr, nullptr, F(84), F(85), F(86), F(87), out, nullptr, nullptr, N);
  mlp2_wmma_kernel<64, 0, 64, 8><<<(M + 63) / 64, 128, 0, stream>>>(
      hn_cur, nullptr, nullptr, nullptr, F(88), F(89), F(90), F(91),
      out + (long)N * 8, nullptr, nullptr, M);
  mlp2_wmma_kernel<64, 0, 128, 8><<<(NS + 63) / 64, 128, 0, stream>>>(
      vn, nullptr, nullptr, nullptr, F(92), F(93), F(94), F(95),
      out + (long)(N + M) * 8, nullptr, nullptr, NS);
}
